// SelfAttention_30262339567971
// MI455X (gfx1250) — matrix operand order repaired and checked by reading, not compiled, not run on hardware
//
#include <hip/hip_runtime.h>
#include <hip/hip_bf16.h>

// ---------------------------------------------------------------------------
// MI455X (gfx1250) fused causal self-attention, bf16 WMMA pipeline.
//   B=4, T=2048, C=1024, H=16, Dh=64   (M = B*T = 8192)
//   Stage 1: qkv = x @ W_attn + b_attn        -> bf16 Q/K/V [B,H,T,Dh] in ws
//   Stage 2: flash attention per (b,h,qtile)  -> bf16 Y [B,T,C] in ws
//            (K tile staged by the Tensor Data Mover when available)
//   Stage 3: out = Y @ W_proj + b_proj        -> fp32 d_out
//            (A tile staged with global_load_async_to_lds_b128)
// All matmuls: v_wmma_f32_16x16x32_bf16, fp32 accumulate.
// ---------------------------------------------------------------------------

typedef __bf16 bf16;
typedef __attribute__((ext_vector_type(16))) __bf16 bf16x16;
typedef __attribute__((ext_vector_type(8)))  __bf16 bf16x8;
typedef __attribute__((ext_vector_type(4)))  __bf16 bf16x4;
typedef __attribute__((ext_vector_type(8)))  float  f32x8;
typedef __attribute__((ext_vector_type(4)))  float  f32x4;
typedef __attribute__((ext_vector_type(4)))  unsigned int u32x4;
typedef __attribute__((ext_vector_type(8)))  int    i32x8;
typedef __attribute__((ext_vector_type(4)))  int    i32x4;

union Frag16 { bf16x16 v; bf16x8 h[2]; };

#define WMMA_BF16(A, B, C) \
  __builtin_amdgcn_wmma_f32_16x16x32_bf16(false, (A), false, (B), (short)0, (C), false, false)

// Tensor Data Mover path. This toolchain declares the 6-arg builtin:
//   (u32x4 g0, i32x8 g1, i32x4 g2, i32x4 g3, i32x8 extra, i32 cpol)
#if defined(__AMDGCN__) && __has_builtin(__builtin_amdgcn_tensor_load_to_lds) && \
    __has_builtin(__builtin_amdgcn_s_wait_tensorcnt)
#define HAVE_TDM 1
#else
#define HAVE_TDM 0
#endif

// Async global->LDS copies via inline asm (gfx1250 VFLAT encoding).
#if defined(__gfx1250__)
#define USE_ASYNC_ASM 1
#else
#define USE_ASYNC_ASM 0
#endif

#define NB   4
#define TT   2048
#define CC   1024
#define NH   16
#define DH   64
#define MTOT (NB * TT)   // 8192
#define NQKV (3 * CC)    // 3072

__device__ __forceinline__ f32x8 zero8() {
  f32x8 z = {0.f, 0.f, 0.f, 0.f, 0.f, 0.f, 0.f, 0.f};
  return z;
}

// ---------------------------------------------------------------------------
// Stage 1: QKV projection.  C[m,n] = sum_k x[m,k] * W[k,n] + b[n]
// Workgroup: 256 threads (8 waves), tile 128(M) x 128(N), K-step 32.
// ---------------------------------------------------------------------------
__global__ __launch_bounds__(256) void qkv_gemm(
    const float* __restrict__ X, const float* __restrict__ W,
    const float* __restrict__ bias,
    bf16* __restrict__ Qo, bf16* __restrict__ Ko, bf16* __restrict__ Vo) {
  __shared__ __align__(16) bf16 As[128 * 40];
  __shared__ __align__(16) bf16 Bs[128 * 40];

  const int tid = threadIdx.x;
  const int wid = tid >> 5, lane = tid & 31;
  const int wm = wid & 3, wn = wid >> 2;
  const int rowBase = blockIdx.y * 128;
  const int colBase = blockIdx.x * 128;
  const int n15 = lane & 15;
  const int hi = (lane < 16) ? 0 : 1;

  f32x8 acc[2][4];
#pragma unroll
  for (int i = 0; i < 2; ++i)
#pragma unroll
    for (int j = 0; j < 4; ++j) acc[i][j] = zero8();

  for (int ko = 0; ko < CC; ko += 32) {
    __syncthreads();
    // ---- stage A tile: 128x32 fp32 -> bf16 LDS ----
#pragma unroll
    for (int i = 0; i < 4; ++i) {
      int lin = tid + i * 256;            // float4 index, 1024 total
      int r = lin >> 3;                   // 8 float4 per 32-wide row
      int kc = (lin & 7) << 2;
      f32x4 a = *(const f32x4*)&X[(size_t)(rowBase + r) * CC + ko + kc];
      bf16x4 t;
      t.x = (bf16)a.x; t.y = (bf16)a.y; t.z = (bf16)a.z; t.w = (bf16)a.w;
      *(bf16x4*)&As[r * 40 + kc] = t;
    }
    // ---- stage B tile: 32x128 fp32 -> transposed bf16 LDS [n][k] ----
#pragma unroll
    for (int i = 0; i < 4; ++i) {
      int lin = tid + i * 256;
      int kr = lin >> 5;                  // 32 float4 per 128-wide row
      int nc = (lin & 31) << 2;
      f32x4 w = *(const f32x4*)&W[(size_t)(ko + kr) * NQKV + colBase + nc];
      Bs[(nc + 0) * 40 + kr] = (bf16)w.x;
      Bs[(nc + 1) * 40 + kr] = (bf16)w.y;
      Bs[(nc + 2) * 40 + kr] = (bf16)w.z;
      Bs[(nc + 3) * 40 + kr] = (bf16)w.w;
    }
    if (ko + 32 < CC) {
      __builtin_prefetch(&X[(size_t)(rowBase + (tid >> 1)) * CC + ko + 32], 0, 0);
      __builtin_prefetch(&W[(size_t)(ko + 32 + (tid >> 3)) * NQKV + colBase], 0, 0);
    }
    __syncthreads();

    // ---- fragments + 8 WMMA ----
    Frag16 af[2], bfr[4];
    const int aoff = hi ? 8 : 0;          // A: lanes<16 get K {0-7,16-23}
    const int boff = hi ? 8 : 0;          // B: lanes<16 get K {0-7,16-23}
#pragma unroll
    for (int fm = 0; fm < 2; ++fm) {
      int row = wm * 32 + fm * 16 + n15;
      af[fm].h[0] = *(const bf16x8*)&As[row * 40 + aoff];
      af[fm].h[1] = *(const bf16x8*)&As[row * 40 + aoff + 16];
    }
#pragma unroll
    for (int fn = 0; fn < 4; ++fn) {
      int col = wn * 64 + fn * 16 + n15;
      bfr[fn].h[0] = *(const bf16x8*)&Bs[col * 40 + boff];
      bfr[fn].h[1] = *(const bf16x8*)&Bs[col * 40 + boff + 16];
    }
#pragma unroll
    for (int fm = 0; fm < 2; ++fm)
#pragma unroll
      for (int fn = 0; fn < 4; ++fn)
        acc[fm][fn] = WMMA_BF16(af[fm].v, bfr[fn].v, acc[fm][fn]);
  }

  // ---- epilogue: + bias, scatter bf16 into Q/K/V [B,H,T,Dh] ----
#pragma unroll
  for (int fm = 0; fm < 2; ++fm)
#pragma unroll
    for (int fn = 0; fn < 4; ++fn) {
      int ncol = colBase + wn * 64 + fn * 16 + n15;
      float bv = bias[ncol];
      int sec = ncol >> 10;               // 0=Q 1=K 2=V
      int rem = ncol & 1023;
      int h = rem >> 6, d = rem & 63;
      bf16* dst = (sec == 0) ? Qo : ((sec == 1) ? Ko : Vo);
#pragma unroll
      for (int r = 0; r < 8; ++r) {
        int mrow = rowBase + wm * 32 + fm * 16 + r + hi * 8;
        int bidx = mrow >> 11, t = mrow & 2047;
        float val = acc[fm][fn][r] + bv;
        dst[(((size_t)(bidx * NH + h) * TT + t) << 6) + d] = (bf16)val;
      }
    }
}

// ---------------------------------------------------------------------------
// Stage 2: flash attention.  grid = (T/64, B*H), block = 128 (4 waves).
// K tile (64x64 bf16, LDS rows padded 128B+16B) staged by the TDM; V tile
// transposed manually; online softmax in fp32; P re-laid out through
// per-wave LDS for the PV WMMA.
// ---------------------------------------------------------------------------
__global__ __launch_bounds__(128) void attn(
    const bf16* __restrict__ Qg, const bf16* __restrict__ Kg,
    const bf16* __restrict__ Vg, bf16* __restrict__ Y) {
  __shared__ __align__(16) bf16 Ks[64 * 72];     // [key][dim], 144B rows
  __shared__ __align__(16) bf16 Vt[64 * 72];     // [dim][key]
  __shared__ __align__(16) bf16 Pl[4 * 16 * 72]; // per-wave [m][key]

  const int tid = threadIdx.x, wid = tid >> 5, lane = tid & 31;
  const int tile = blockIdx.x;                   // query tile (64 rows)
  const int bh = blockIdx.y;
  const int b = bh >> 4, h = bh & 15;
  const size_t base = (size_t)bh * TT * DH;
  const int n15 = lane & 15;
  const int hi = (lane < 16) ? 0 : 1;
  const int aoff = hi ? 8 : 0;
  const int boff = hi ? 8 : 0;

#if HAVE_TDM
  // TDM descriptor group 1 (invariant): 2-D 64x64 tile of 2-byte elements,
  // tensor_dim0=64, tensor_dim1=64, stride0=64, LDS pad: every 32 DWORDs
  // (128B row) insert 4 DWORDs (16B) -> matches stride-72 bf16 rows.
  i32x8 tdmG1;
  tdmG1[0] = (int)((1u << 16) |          // data_size = 2 bytes
                   (1u << 20) |          // pad_enable
                   (4u << 22) |          // pad_interval: 32 DWORDs
                   (3u << 25));          // pad_amount:   4 DWORDs
  tdmG1[1] = (int)(64u << 16);           // tensor_dim0 = 64 (bits 79:48 lo)
  tdmG1[2] = (int)(64u << 16);           // tensor_dim1 = 64 (bits 95:80)
  tdmG1[3] = (int)(64u << 16);           // tile_dim0   = 64 (bits 127:112)
  tdmG1[4] = 64;                         // tile_dim1   = 64 (bits 143:128)
  tdmG1[5] = 64;                         // tensor_dim0_stride = 64
  tdmG1[6] = 0;
  tdmG1[7] = 0;
  i32x4 tdmZ4 = {0, 0, 0, 0};
  i32x8 tdmZ8 = {0, 0, 0, 0, 0, 0, 0, 0};
  const unsigned ldsK = (unsigned)(size_t)&Ks[0];
#endif

  // Q fragments for this wave's 16 rows (2 K-steps over Dh=64)
  Frag16 qf[2];
  {
    const bf16* qp = Qg + base + (size_t)(tile * 64 + wid * 16 + n15) * DH;
#pragma unroll
    for (int ks = 0; ks < 2; ++ks) {
      qf[ks].h[0] = *(const bf16x8*)&qp[ks * 32 + aoff];
      qf[ks].h[1] = *(const bf16x8*)&qp[ks * 32 + aoff + 16];
    }
  }

  f32x8 of[4];
#pragma unroll
  for (int g = 0; g < 4; ++g) of[g] = zero8();
  float mrow[8], lrow[8];
#pragma unroll
  for (int r = 0; r < 8; ++r) { mrow[r] = -3.0e38f; lrow[r] = 0.f; }

  const int nkt = tile + 1;  // causal
  for (int kt = 0; kt < nkt; ++kt) {
    __syncthreads();
#if HAVE_TDM
    if (wid == 0) {           // one wave drives the tensor DMA for the K tile
      size_t gaddr = (size_t)(Kg + base + (size_t)kt * 64 * DH);
      u32x4 g0;
      g0[0] = 1u;                                        // count=1
      g0[1] = ldsK;                                      // lds_addr
      g0[2] = (unsigned)gaddr;                           // global_addr lo
      g0[3] = ((unsigned)(gaddr >> 32) & 0x01FFFFFFu) |  // global_addr hi
              (2u << 30);                                // type = image
      __builtin_amdgcn_tensor_load_to_lds(g0, tdmG1, tdmZ4, tdmZ4, tdmZ8, 0);
    }
#endif
    // ---- stage V transposed (and K directly when no TDM) ----
#pragma unroll
    for (int i = 0; i < 4; ++i) {
      int lin = tid + i * 128;                 // 512 chunks of 8 bf16
      int r = lin >> 3;                        // key row 0..63
      int cc = (lin & 7) << 3;                 // dim chunk
#if !HAVE_TDM
      *(bf16x8*)&Ks[r * 72 + cc] =
          *(const bf16x8*)&Kg[base + (size_t)(kt * 64 + r) * DH + cc];
#endif
      bf16x8 vv = *(const bf16x8*)&Vg[base + (size_t)(kt * 64 + r) * DH + cc];
#pragma unroll
      for (int j = 0; j < 8; ++j) Vt[(cc + j) * 72 + r] = vv[j];
    }
#if HAVE_TDM
    if (wid == 0) __builtin_amdgcn_s_wait_tensorcnt(0);
#endif
    __syncthreads();

    // ---- S = (Q K^T) * scale ----
    f32x8 sf[4];
#pragma unroll
    for (int f = 0; f < 4; ++f) {
      sf[f] = zero8();
      int key = f * 16 + n15;
#pragma unroll
      for (int ks = 0; ks < 2; ++ks) {
        Frag16 kb;
        kb.h[0] = *(const bf16x8*)&Ks[key * 72 + ks * 32 + boff];
        kb.h[1] = *(const bf16x8*)&Ks[key * 72 + ks * 32 + boff + 16];
        sf[f] = WMMA_BF16(qf[ks].v, kb.v, sf[f]);
      }
    }
    const bool diag = (kt == tile);
#pragma unroll
    for (int f = 0; f < 4; ++f) {
      int key = kt * 64 + f * 16 + n15;
#pragma unroll
      for (int r = 0; r < 8; ++r) {
        float s = sf[f][r] * 0.125f;         // Dh^-0.5
        int q = tile * 64 + wid * 16 + r + hi * 8;
        if (diag && key > q) s = -3.0e38f;
        sf[f][r] = s;
      }
    }

    // ---- online softmax (row stats via intra-16-lane shuffles) ----
#pragma unroll
    for (int r = 0; r < 8; ++r) {
      float mx = fmaxf(fmaxf(sf[0][r], sf[1][r]), fmaxf(sf[2][r], sf[3][r]));
#pragma unroll
      for (int d = 1; d < 16; d <<= 1) mx = fmaxf(mx, __shfl_xor(mx, d, 32));
      float mnew = fmaxf(mrow[r], mx);
      float corr = __expf(mrow[r] - mnew);
      mrow[r] = mnew;
      float rs = 0.f;
#pragma unroll
      for (int f = 0; f < 4; ++f) {
        float p = __expf(sf[f][r] - mnew);
        sf[f][r] = p;
        rs += p;
      }
#pragma unroll
      for (int d = 1; d < 16; d <<= 1) rs += __shfl_xor(rs, d, 32);
      lrow[r] = lrow[r] * corr + rs;
#pragma unroll
      for (int g = 0; g < 4; ++g) of[g][r] *= corr;
    }

    // ---- P (C-layout) -> per-wave LDS [m][key] -> A fragments ----
    bf16* pw = &Pl[wid * 16 * 72];
#pragma unroll
    for (int f = 0; f < 4; ++f) {
      int key = f * 16 + n15;
#pragma unroll
      for (int r = 0; r < 8; ++r) pw[(r + hi * 8) * 72 + key] = (bf16)sf[f][r];
    }
    // ---- O += P @ V ----
#pragma unroll
    for (int g = 0; g < 4; ++g) {
      int dh = g * 16 + n15;
#pragma unroll
      for (int ks = 0; ks < 2; ++ks) {
        Frag16 pa, vb;
        pa.h[0] = *(const bf16x8*)&pw[n15 * 72 + ks * 32 + aoff];
        pa.h[1] = *(const bf16x8*)&pw[n15 * 72 + ks * 32 + aoff + 16];
        vb.h[0] = *(const bf16x8*)&Vt[dh * 72 + ks * 32 + boff];
        vb.h[1] = *(const bf16x8*)&Vt[dh * 72 + ks * 32 + boff + 16];
        of[g] = WMMA_BF16(pa.v, vb.v, of[g]);
      }
    }
  }

  // ---- epilogue: O /= l, bf16 Y in [B,T,C] ----
#pragma unroll
  for (int g = 0; g < 4; ++g) {
    int col = h * DH + g * 16 + n15;
#pragma unroll
    for (int r = 0; r < 8; ++r) {
      int q = tile * 64 + wid * 16 + r + hi * 8;
      float val = of[g][r] / lrow[r];
      Y[((size_t)(b * TT + q) << 10) + col] = (bf16)val;
    }
  }
}

// ---------------------------------------------------------------------------
// Stage 3: output projection.  out[m,n] = sum_k Y[m,k]*Wp[k,n] + bp[n] (fp32)
// A tile is bf16 in workspace: staged with async global->LDS copies.
// ---------------------------------------------------------------------------
__global__ __launch_bounds__(256) void proj_gemm(
    const bf16* __restrict__ Yb, const float* __restrict__ W,
    const float* __restrict__ bias, float* __restrict__ Out) {
  __shared__ __align__(16) bf16 As[128 * 40];
  __shared__ __align__(16) bf16 Bs[128 * 40];

  const int tid = threadIdx.x;
  const int wid = tid >> 5, lane = tid & 31;
  const int wm = wid & 3, wn = wid >> 2;
  const int rowBase = blockIdx.y * 128;
  const int colBase = blockIdx.x * 128;
  const int n15 = lane & 15;
  const int hi = (lane < 16) ? 0 : 1;

  f32x8 acc[2][4];
#pragma unroll
  for (int i = 0; i < 2; ++i)
#pragma unroll
    for (int j = 0; j < 4; ++j) acc[i][j] = zero8();

  for (int ko = 0; ko < CC; ko += 32) {
    __syncthreads();
    // ---- A tile: 128x32 bf16, async copy global -> LDS ----
#if USE_ASYNC_ASM
#pragma unroll
    for (int i = 0; i < 2; ++i) {
      int lin = tid + i * 256;                // bf16x8 chunk id, 512 total
      int r = lin >> 2;                       // 4 chunks per 32-wide row
      int kc = (lin & 3) << 3;
      unsigned lo = (unsigned)(size_t)&As[r * 40 + kc];
      unsigned long long ga =
          (unsigned long long)(size_t)&Yb[(size_t)(rowBase + r) * CC + ko + kc];
      asm volatile("global_load_async_to_lds_b128 %0, %1, off"
                   :: "v"(lo), "v"(ga) : "memory");
    }
#else
#pragma unroll
    for (int i = 0; i < 2; ++i) {
      int lin = tid + i * 256;
      int r = lin >> 2;
      int kc = (lin & 3) << 3;
      *(bf16x8*)&As[r * 40 + kc] =
          *(const bf16x8*)&Yb[(size_t)(rowBase + r) * CC + ko + kc];
    }
#endif
    // ---- B tile: 32x128 fp32 -> transposed bf16 [n][k] ----
#pragma unroll
    for (int i = 0; i < 4; ++i) {
      int lin = tid + i * 256;
      int kr = lin >> 5;
      int nc = (lin & 31) << 2;
      f32x4 w = *(const f32x4*)&W[(size_t)(ko + kr) * CC + colBase + nc];
      Bs[(nc + 0) * 40 + kr] = (bf16)w.x;
      Bs[(nc + 1) * 40 + kr] = (bf16)w.y;
      Bs[(nc + 2) * 40 + kr] = (bf16)w.z;
      Bs[(nc + 3) * 40 + kr] = (bf16)w.w;
    }
    if (ko + 32 < CC) {
      __builtin_prefetch(&W[(size_t)(ko + 32 + (tid >> 3)) * CC + colBase], 0, 0);
    }
#if USE_ASYNC_ASM
    asm volatile("s_wait_asynccnt 0x0" ::: "memory");
#endif
    __syncthreads();

    Frag16 af[2], bfr[4];
    const int aoff = hi ? 8 : 0;
    const int boff = hi ? 8 : 0;
#pragma unroll
    for (int fm = 0; fm < 2; ++fm) {
      int row = wm * 32 + fm * 16 + n15;
      af[fm].h[0] = *(const bf16x8*)&As[row * 40 + aoff];
      af[fm].h[1] = *(const bf16x8*)&As[row * 40 + aoff + 16];
    }
#pragma unroll
    for (int fn = 0; fn < 4; ++fn) {
      int col = wn * 64 + fn * 16 + n15;
      bfr[fn].h[0] = *(const bf16x8*)&Bs[col * 40 + boff];
      bfr[fn].h[1] = *(const bf16x8*)&Bs[col * 40 + boff + 16];
    }
#pragma unroll
    for (int fm = 0; fm < 2; ++fm)
#pragma unroll
      for (int fn = 0; fn < 4; ++fn)
        acc[fm][fn] = WMMA_BF16(af[fm].v, bfr[fn].v, acc[fm][fn]);
  }

#pragma unroll
  for (int fm = 0; fm < 2; ++fm)
#pragma unroll
    for (int fn = 0; fn < 4; ++fn) {
      int ncol = colBase + wn * 64 + fn * 16 + n15;
      float bv = bias[ncol];
#pragma unroll
      for (int r = 0; r < 8; ++r) {
        int mrow = rowBase + wm * 32 + fm * 16 + r + hi * 8;
        Out[(size_t)mrow * CC + ncol] = acc[fm][fn][r] + bv;
      }
    }
}

// ---------------------------------------------------------------------------
extern "C" void kernel_launch(void* const* d_in, const int* in_sizes, int n_in,
                              void* d_out, int out_size, void* d_ws, size_t ws_size,
                              hipStream_t stream) {
  const float* x      = (const float*)d_in[0];
  const float* W_attn = (const float*)d_in[1];
  const float* b_attn = (const float*)d_in[2];
  const float* W_proj = (const float*)d_in[3];
  const float* b_proj = (const float*)d_in[4];
  float* out = (float*)d_out;

  // workspace: Q, K, V [B,H,T,Dh] bf16 + Y [B,T,C] bf16  = 64 MB
  const size_t nelem = (size_t)NB * NH * TT * DH;  // 8388608
  bf16* q = (bf16*)d_ws;
  bf16* k = q + nelem;
  bf16* v = k + nelem;
  bf16* y = v + nelem;

  qkv_gemm<<<dim3(NQKV / 128, MTOT / 128), 256, 0, stream>>>(x, W_attn, b_attn, q, k, v);
  attn<<<dim3(TT / 64, NB * NH), 128, 0, stream>>>(q, k, v, y);
  proj_gemm<<<dim3(CC / 128, MTOT / 128), 256, 0, stream>>>(y, W_proj, b_proj, out);
}
